// SparseMultiHeadAttention_16664473108971
// MI455X (gfx1250) — matrix operand order repaired and checked by reading, not compiled, not run on hardware
//
#include <hip/hip_runtime.h>
#include <hip/hip_bf16.h>
#include <math.h>

#define E_DIM 1024
#define H_NUM 16
#define D_DIM 64
#define WIN_SZ 256
#define STRIDE_W 128
#define B_NUM 2
#define L_SEQ 4097
#define LC (L_SEQ - 1)
#define NWIN 31
#define M_ROWS (B_NUM * L_SEQ)   /* 8194 */
#define ATT_SCALE 0.125f          /* 1/sqrt(64) */

typedef __attribute__((ext_vector_type(16))) __bf16 v16bf;
typedef __attribute__((ext_vector_type(8)))  float  v8f;

union FragA { v16bf v; unsigned int u[8]; };
union FragC { v8f   v; float        f[8]; };

__device__ __forceinline__ unsigned short f2bf(float x) {
  union { float f; unsigned int u; } c; c.f = x;
  unsigned int u = c.u + 0x7FFFu + ((c.u >> 16) & 1u);   // RNE
  return (unsigned short)(u >> 16);
}
__device__ __forceinline__ float bf2f(unsigned short h) {
  union { unsigned int u; float f; } c; c.u = ((unsigned int)h) << 16; return c.f;
}
__device__ __forceinline__ unsigned int pack2(unsigned short lo, unsigned short hi) {
  return (unsigned int)lo | ((unsigned int)hi << 16);
}
// WMMA bf16 A-matrix 16x32: VGPR i on lane-half -> packed K pair starting at:
__device__ __forceinline__ int a_k0(int i, int half) {
  return ((i < 4) ? (2 * i) : (16 + 2 * (i - 4))) + 8 * half;
}
// WMMA bf16 B-matrix 32x16: VGPR i on lane-half -> packed K pair starting at (same as A):
__device__ __forceinline__ int b_k0(int i, int half) {
  return ((i < 4) ? (2 * i) : (16 + 2 * (i - 4))) + 8 * half;
}

// ---------------------------------------------------------------------------
// Kernel 1: fused QKV projection GEMM.
// C[8194 x 3072] = X[8194 x 1024] * [Wq|Wk|Wv], + bias, scatter to
// head-major bf16 q/k/v buffers [b][h][s][d].
// Block tile 128x64x32, 8 waves (4x2), wave tile 32x32 (2x2 WMMA 16x16x32).
// Sized so accumulators + fragments + staging fit a ~128-VGPR budget.
// ---------------------------------------------------------------------------
#define BM 128
#define BN 64
#define BK 32
#define LDA 34

__global__ __launch_bounds__(256) __attribute__((amdgpu_waves_per_eu(1)))
void qkv_gemm(
    const float* __restrict__ X,
    const float* __restrict__ Wq, const float* __restrict__ Wk,
    const float* __restrict__ Wv,
    const float* __restrict__ bq, const float* __restrict__ bk,
    const float* __restrict__ bv,
    unsigned short* __restrict__ qb, unsigned short* __restrict__ kb,
    unsigned short* __restrict__ vb)
{
  __shared__ __align__(8) unsigned short As[BM * LDA];
  __shared__ __align__(8) unsigned short Bs[BN * LDA];

  const int tid  = threadIdx.x;
  const int lane = tid & 31;
  const int wave = tid >> 5;
  const int half = lane >> 4;
  const int l16  = lane & 15;
  const int waveM = (wave & 3) * 32;   // 0,32,64,96
  const int waveN = (wave >> 2) * 32;  // 0,32

  const int m0  = blockIdx.x * BM;
  const int nb  = blockIdx.y;          // 0..47
  const int sel = nb >> 4;             // 0:q 1:k 2:v (64 | 1024 -> no straddle)
  const int n0  = (nb & 15) * BN;
  const float* W    = (sel == 0) ? Wq : (sel == 1) ? Wk : Wv;
  const float* bias = (sel == 0) ? bq : (sel == 1) ? bk : bv;

  FragC acc[2][2];
  for (int mt = 0; mt < 2; ++mt)
    for (int nt = 0; nt < 2; ++nt)
      for (int i = 0; i < 8; ++i) acc[mt][nt].f[i] = 0.f;

  for (int kt = 0; kt < E_DIM / BK; ++kt) {
    for (int idx = tid; idx < BM * BK; idx += 256) {
      int r = idx >> 5, c = idx & 31;
      int gm = m0 + r;
      float v = (gm < M_ROWS) ? X[(size_t)gm * E_DIM + kt * BK + c] : 0.f;
      As[r * LDA + c] = f2bf(v);
    }
    for (int idx = tid; idx < BN * BK; idx += 256) {
      int n = idx >> 5, c = idx & 31;
      Bs[n * LDA + c] = f2bf(W[(size_t)(kt * BK + c) * E_DIM + n0 + n]);
    }
    __syncthreads();

    FragA af[2];
    for (int mt = 0; mt < 2; ++mt) {
      int mrow = waveM + mt * 16 + l16;
      for (int i = 0; i < 8; ++i)
        af[mt].u[i] = *(const unsigned int*)&As[mrow * LDA + a_k0(i, half)];
    }
    for (int nt = 0; nt < 2; ++nt) {
      FragA bw;
      int nrow = waveN + nt * 16 + l16;
      for (int i = 0; i < 8; ++i)
        bw.u[i] = *(const unsigned int*)&Bs[nrow * LDA + b_k0(i, half)];
      for (int mt = 0; mt < 2; ++mt)
        acc[mt][nt].v = __builtin_amdgcn_wmma_f32_16x16x32_bf16(
            false, af[mt].v, false, bw.v, (short)0, acc[mt][nt].v,
            false, false);
    }
    __syncthreads();
  }

  unsigned short* outb = (sel == 0) ? qb : (sel == 1) ? kb : vb;
  for (int mt = 0; mt < 2; ++mt)
    for (int nt = 0; nt < 2; ++nt) {
      int gn = n0 + waveN + nt * 16 + l16;   // 0..1023 within weight
      float bval = bias[gn];
      int h = gn >> 6, d = gn & 63;
      for (int r = 0; r < 8; ++r) {
        int gm = m0 + waveM + mt * 16 + r + 8 * half;
        if (gm < M_ROWS) {
          int b = gm / L_SEQ, s = gm % L_SEQ;
          outb[(((size_t)b * H_NUM + h) * L_SEQ + s) * D_DIM + d] =
              f2bf(acc[mt][nt].f[r] + bval);
        }
      }
    }
}

// ---------------------------------------------------------------------------
// Kernel 2: global token attention (1 query x 4097 keys per (b,h)). Tiny.
// ---------------------------------------------------------------------------
__global__ __launch_bounds__(256) void global_attn(
    const unsigned short* __restrict__ qb, const unsigned short* __restrict__ kb,
    const unsigned short* __restrict__ vb, unsigned short* __restrict__ attnout)
{
  __shared__ float qsh[D_DIM];
  __shared__ float sc[L_SEQ];
  __shared__ float red[256];
  int b = blockIdx.x >> 4;
  int h = blockIdx.x & 15;
  int tid = threadIdx.x;
  const size_t headOff = ((size_t)b * H_NUM + h) * L_SEQ * D_DIM;
  if (tid < D_DIM) qsh[tid] = bf2f(qb[headOff + tid]);   // s = 0
  __syncthreads();
  float lmax = -INFINITY;
  for (int s = tid; s < L_SEQ; s += 256) {
    const unsigned short* kr = kb + headOff + (size_t)s * D_DIM;
    float dot = 0.f;
    for (int d = 0; d < D_DIM; ++d) dot += qsh[d] * bf2f(kr[d]);
    dot *= ATT_SCALE;
    sc[s] = dot;
    lmax = fmaxf(lmax, dot);
  }
  red[tid] = lmax; __syncthreads();
  for (int off = 128; off > 0; off >>= 1) {
    if (tid < off) red[tid] = fmaxf(red[tid], red[tid + off]);
    __syncthreads();
  }
  float gmax = red[0]; __syncthreads();
  float lsum = 0.f;
  for (int s = tid; s < L_SEQ; s += 256) {
    float e = __expf(sc[s] - gmax);
    sc[s] = e; lsum += e;
  }
  red[tid] = lsum; __syncthreads();
  for (int off = 128; off > 0; off >>= 1) {
    if (tid < off) red[tid] += red[tid + off];
    __syncthreads();
  }
  float inv = 1.f / red[0];
  if (tid < D_DIM) {
    float accv = 0.f;
    for (int s = 0; s < L_SEQ; ++s)
      accv += sc[s] * bf2f(vb[headOff + (size_t)s * D_DIM + tid]);
    attnout[((size_t)b * L_SEQ) * E_DIM + h * D_DIM + tid] = f2bf(accv * inv);
  }
}

// ---------------------------------------------------------------------------
// Kernel 3: windowed attention. Two blocks per (b, window, head), each block
// covers 128 queries with 8 waves x 16 queries. Flash-style over 4 key-chunks
// of 64; all matmuls via WMMA bf16. Q fragments reloaded per k-step (L2 hit)
// to keep peak VGPR use ~110.
// ---------------------------------------------------------------------------
#define PLD 66
__global__ __launch_bounds__(256) __attribute__((amdgpu_waves_per_eu(1)))
void window_attn(
    const unsigned short* __restrict__ qb, const unsigned short* __restrict__ kb,
    const unsigned short* __restrict__ vb, unsigned short* __restrict__ winctx)
{
  __shared__ __align__(8) unsigned short Kt[64 * PLD];
  __shared__ __align__(8) unsigned short Vt[64 * PLD];
  __shared__ __align__(8) unsigned short Pst[8 * 16 * PLD];

  const int tid  = threadIdx.x;
  const int lane = tid & 31;
  const int wave = tid >> 5;
  const int half = lane >> 4;
  const int l16  = lane & 15;

  int t = blockIdx.x;
  const int qh = t & 1;  t >>= 1;      // query half: 0 or 1 (128 queries each)
  const int h  = t % H_NUM; t /= H_NUM;
  const int w  = t % NWIN;
  const int b  = t / NWIN;

  const size_t headOff = ((size_t)b * H_NUM + h) * L_SEQ * D_DIM;
  const int s0 = 1 + w * STRIDE_W;           // skip global token
  const int qbase = qh * 128 + wave * 16;    // this wave's 16 queries

  FragC O[4];
  float rm[8], rs[8];
  for (int dt = 0; dt < 4; ++dt)
    for (int i = 0; i < 8; ++i) O[dt].f[i] = 0.f;
  for (int r = 0; r < 8; ++r) { rm[r] = -INFINITY; rs[r] = 0.f; }

  const unsigned int* qp = (const unsigned int*)(
      qb + headOff + (size_t)(s0 + qbase + l16) * D_DIM);

  for (int kc = 0; kc < 4; ++kc) {
    __syncthreads();
    // Stage 64-key K/V chunk into LDS.
    {
      const unsigned int* kg = (const unsigned int*)(kb + headOff + (size_t)(s0 + kc * 64) * D_DIM);
      const unsigned int* vg = (const unsigned int*)(vb + headOff + (size_t)(s0 + kc * 64) * D_DIM);
      for (int ui = tid; ui < 64 * 32; ui += 256) {
        int r = ui >> 5;
        int c = (ui & 31) << 1;
        *(unsigned int*)&Kt[r * PLD + c] = kg[ui];
        *(unsigned int*)&Vt[r * PLD + c] = vg[ui];
      }
    }
    __syncthreads();

    // S = Q * K^T (f32 accum); Q fragment reloaded per k-step, one K fragment
    // live at a time.
    FragC S[4];
    for (int nt = 0; nt < 4; ++nt)
      for (int i = 0; i < 8; ++i) S[nt].f[i] = 0.f;

    for (int ds = 0; ds < 2; ++ds) {
      FragA qa;
      for (int i = 0; i < 8; ++i)
        qa.u[i] = qp[(ds * 32 + a_k0(i, half)) >> 1];
      for (int nt = 0; nt < 4; ++nt) {
        FragA bk_;
        int key = nt * 16 + l16;
        for (int i = 0; i < 8; ++i)
          bk_.u[i] = *(const unsigned int*)&Kt[key * PLD + ds * 32 + b_k0(i, half)];
        S[nt].v = __builtin_amdgcn_wmma_f32_16x16x32_bf16(
            false, qa.v, false, bk_.v, (short)0, S[nt].v, false, false);
      }
    }

    // Online softmax update. Row M = r + 8*half lives on one 16-lane group.
    for (int r = 0; r < 8; ++r) {
      float cmax = -INFINITY;
      for (int nt = 0; nt < 4; ++nt)
        cmax = fmaxf(cmax, S[nt].f[r] * ATT_SCALE);
      for (int m = 1; m < 16; m <<= 1)
        cmax = fmaxf(cmax, __shfl_xor(cmax, m, 32));
      float mnew = fmaxf(rm[r], cmax);
      float corr = __expf(rm[r] - mnew);
      rm[r] = mnew;
      float rsum = 0.f;
      for (int nt = 0; nt < 4; ++nt) {
        float p = __expf(S[nt].f[r] * ATT_SCALE - mnew);
        S[nt].f[r] = p;
        rsum += p;
      }
      for (int m = 1; m < 16; m <<= 1)
        rsum += __shfl_xor(rsum, m, 32);
      rs[r] = rs[r] * corr + rsum;
      for (int dt = 0; dt < 4; ++dt) O[dt].f[r] *= corr;
    }

    // Stage P (bf16) to wave-private LDS, C-layout -> memory.
    unsigned short* P = &Pst[wave * 16 * PLD];
    for (int nt = 0; nt < 4; ++nt)
      for (int r = 0; r < 8; ++r)
        P[(r + 8 * half) * PLD + nt * 16 + l16] = f2bf(S[nt].f[r]);
    asm volatile("s_wait_dscnt 0" ::: "memory");   // same-wave LDS RAW

    // O += P * V; one P fragment, stream one V fragment per dt.
    for (int ks = 0; ks < 2; ++ks) {
      FragA pa;
      for (int i = 0; i < 8; ++i)
        pa.u[i] = *(const unsigned int*)&P[l16 * PLD + ks * 32 + a_k0(i, half)];
      for (int dt = 0; dt < 4; ++dt) {
        FragA bv_;
        int dcol = dt * 16 + l16;
        for (int i = 0; i < 8; ++i) {
          int k0 = ks * 32 + b_k0(i, half);
          bv_.u[i] = pack2(Vt[k0 * PLD + dcol], Vt[(k0 + 1) * PLD + dcol]);
        }
        O[dt].v = __builtin_amdgcn_wmma_f32_16x16x32_bf16(
            false, pa.v, false, bv_.v, (short)0, O[dt].v, false, false);
      }
    }
  }

  // Normalize and write per-window context [b][w][h][q][d] bf16.
  unsigned short* wc = winctx + (((size_t)b * NWIN + w) * H_NUM + h) * WIN_SZ * D_DIM;
  for (int dt = 0; dt < 4; ++dt)
    for (int r = 0; r < 8; ++r) {
      int qr = qbase + r + 8 * half;
      int dc = dt * 16 + l16;
      wc[(size_t)qr * D_DIM + dc] = f2bf(O[dt].f[r] / rs[r]);
    }
}

// ---------------------------------------------------------------------------
// Kernel 4: average overlapping window contexts into the attention matrix.
// ---------------------------------------------------------------------------
__global__ void gather_local(const unsigned short* __restrict__ winctx,
                             unsigned short* __restrict__ attnout)
{
  size_t idx = (size_t)blockIdx.x * blockDim.x + threadIdx.x;
  const size_t total = (size_t)B_NUM * LC * E_DIM;
  if (idx >= total) return;
  int e = (int)(idx % E_DIM);
  size_t t = idx / E_DIM;
  int p = (int)(t % LC);
  int b = (int)(t / LC);
  int h = e >> 6, d = e & 63;
  int c = p >> 7;            // 0..31
  float sum = 0.f; int cnt = 0;
  if (c <= NWIN - 1) {       // window c, offset p%128
    sum += bf2f(winctx[((((size_t)b * NWIN + c) * H_NUM + h) * WIN_SZ + (p & 127)) * D_DIM + d]);
    ++cnt;
  }
  if (c >= 1) {              // window c-1, offset p%128 + 128
    sum += bf2f(winctx[((((size_t)b * NWIN + (c - 1)) * H_NUM + h) * WIN_SZ + (p & 127) + 128) * D_DIM + d]);
    ++cnt;
  }
  attnout[((size_t)b * L_SEQ + 1 + p) * E_DIM + e] = f2bf(sum / (float)cnt);
}

// ---------------------------------------------------------------------------
// Kernel 5: output projection. out[8194x1024] = Attn(bf16) * Wo + bo (f32).
// ---------------------------------------------------------------------------
__global__ __launch_bounds__(256) __attribute__((amdgpu_waves_per_eu(1)))
void out_gemm(
    const unsigned short* __restrict__ Abf, const float* __restrict__ Wo,
    const float* __restrict__ bo, float* __restrict__ out)
{
  __shared__ __align__(8) unsigned short As[BM * LDA];
  __shared__ __align__(8) unsigned short Bs[BN * LDA];

  const int tid  = threadIdx.x;
  const int lane = tid & 31;
  const int wave = tid >> 5;
  const int half = lane >> 4;
  const int l16  = lane & 15;
  const int waveM = (wave & 3) * 32;
  const int waveN = (wave >> 2) * 32;
  const int m0 = blockIdx.x * BM;
  const int n0 = blockIdx.y * BN;

  FragC acc[2][2];
  for (int mt = 0; mt < 2; ++mt)
    for (int nt = 0; nt < 2; ++nt)
      for (int i = 0; i < 8; ++i) acc[mt][nt].f[i] = 0.f;

  for (int kt = 0; kt < E_DIM / BK; ++kt) {
    for (int idx = tid; idx < BM * BK; idx += 256) {
      int r = idx >> 5, c = idx & 31;
      int gm = m0 + r;
      As[r * LDA + c] = (gm < M_ROWS)
          ? Abf[(size_t)gm * E_DIM + kt * BK + c] : (unsigned short)0;
    }
    for (int idx = tid; idx < BN * BK; idx += 256) {
      int n = idx >> 5, c = idx & 31;
      Bs[n * LDA + c] = f2bf(Wo[(size_t)(kt * BK + c) * E_DIM + n0 + n]);
    }
    __syncthreads();

    FragA af[2];
    for (int mt = 0; mt < 2; ++mt) {
      int mrow = waveM + mt * 16 + l16;
      for (int i = 0; i < 8; ++i)
        af[mt].u[i] = *(const unsigned int*)&As[mrow * LDA + a_k0(i, half)];
    }
    for (int nt = 0; nt < 2; ++nt) {
      FragA bw;
      int nrow = waveN + nt * 16 + l16;
      for (int i = 0; i < 8; ++i)
        bw.u[i] = *(const unsigned int*)&Bs[nrow * LDA + b_k0(i, half)];
      for (int mt = 0; mt < 2; ++mt)
        acc[mt][nt].v = __builtin_amdgcn_wmma_f32_16x16x32_bf16(
            false, af[mt].v, false, bw.v, (short)0, acc[mt][nt].v,
            false, false);
    }
    __syncthreads();
  }

  for (int mt = 0; mt < 2; ++mt)
    for (int nt = 0; nt < 2; ++nt) {
      int gn = n0 + waveN + nt * 16 + l16;
      float bval = bo[gn];
      for (int r = 0; r < 8; ++r) {
        int gm = m0 + waveM + mt * 16 + r + 8 * half;
        if (gm < M_ROWS)
          out[(size_t)gm * E_DIM + gn] = acc[mt][nt].f[r] + bval;
      }
    }
}

// ---------------------------------------------------------------------------
extern "C" void kernel_launch(void* const* d_in, const int* in_sizes, int n_in,
                              void* d_out, int out_size, void* d_ws, size_t ws_size,
                              hipStream_t stream) {
  const float* X  = (const float*)d_in[0];
  const float* Wq = (const float*)d_in[1];
  const float* bq = (const float*)d_in[2];
  const float* Wk = (const float*)d_in[3];
  const float* bk = (const float*)d_in[4];
  const float* Wv = (const float*)d_in[5];
  const float* bv = (const float*)d_in[6];
  const float* Wo = (const float*)d_in[7];
  const float* bo = (const float*)d_in[8];
  float* out = (float*)d_out;

  // Workspace carve (bf16 everywhere): ~95 MB total.
  const size_t qkvBytes = ((size_t)B_NUM * H_NUM * L_SEQ * D_DIM * 2 + 255) & ~(size_t)255;
  const size_t wcBytes  = ((size_t)B_NUM * NWIN * H_NUM * WIN_SZ * D_DIM * 2 + 255) & ~(size_t)255;
  const size_t aoBytes  = ((size_t)B_NUM * L_SEQ * E_DIM * 2 + 255) & ~(size_t)255;
  char* ws = (char*)d_ws;
  unsigned short* qb      = (unsigned short*)ws;              ws += qkvBytes;
  unsigned short* kb      = (unsigned short*)ws;              ws += qkvBytes;
  unsigned short* vb      = (unsigned short*)ws;              ws += qkvBytes;
  unsigned short* winctx  = (unsigned short*)ws;              ws += wcBytes;
  unsigned short* attnout = (unsigned short*)ws;              ws += aoBytes;
  (void)ws_size; (void)in_sizes; (void)n_in; (void)out_size;

  qkv_gemm<<<dim3((M_ROWS + BM - 1) / BM, 3 * E_DIM / BN), 256, 0, stream>>>(
      X, Wq, Wk, Wv, bq, bk, bv, qb, kb, vb);
  global_attn<<<dim3(B_NUM * H_NUM), 256, 0, stream>>>(qb, kb, vb, attnout);
  window_attn<<<dim3(B_NUM * NWIN * H_NUM * 2), 256, 0, stream>>>(qb, kb, vb, winctx);
  gather_local<<<dim3((unsigned)(((size_t)B_NUM * LC * E_DIM + 255) / 256)), 256, 0, stream>>>(
      winctx, attnout);
  out_gemm<<<dim3((M_ROWS + BM - 1) / BM, E_DIM / BN), 256, 0, stream>>>(
      attnout, Wo, bo, out);
}
